// Block_66065186947619
// MI455X (gfx1250) — matrix operand order repaired and checked by reading, not compiled, not run on hardware
//
#include <hip/hip_runtime.h>
#include <hip/hip_bf16.h>
#include <cstddef>

// ---------------------------------------------------------------------------
// AdaBin ConvNeXt-ish block, fused for gfx1250 (MI455X):
//   K0: quantize weights (dw 7x7, pw1, pw2) -> ws (pw weights fp16, [N][K])
//   K1: depthwise 7x7 binary conv + bn1 + residual + pw1-activation-quantize
//       -> h_q fp16, layout [M=32768][K=384]
//   K2: GEMM1 (M=32768,K=384,N=1536)  f16 WMMA, epilogue bn2+maxout+pw2-quant
//   K3: GEMM2 (M=32768,K=1536,N=384)  f16 WMMA, epilogue bn3 + input residual
// GEMM staging uses GLOBAL_LOAD_ASYNC_TO_LDS_B128 with a double-buffered
// pipeline synchronized on ASYNCcnt (fallback: synchronous b128 + ds_store).
// ---------------------------------------------------------------------------

#define DIM   384
#define EXP_  1536
#define BATCH 32
#define HW    1024              // 32*32
#define M_TOT (BATCH * HW)      // 32768
#define BN_EPS 1e-5f

typedef _Float16 v16h __attribute__((ext_vector_type(16)));
typedef _Float16 h8   __attribute__((ext_vector_type(8)));
typedef float    v8f  __attribute__((ext_vector_type(8)));
typedef int      v4i_vs __attribute__((vector_size(16)));   // builtin ptr elem type

#if defined(__HIP_DEVICE_COMPILE__) &&                                    \
    __has_builtin(__builtin_amdgcn_global_load_async_to_lds_b128) &&      \
    __has_builtin(__builtin_amdgcn_s_wait_asynccnt)
#define USE_ASYNC_LDS 1
#else
#define USE_ASYNC_LDS 0
#endif

// LDS row stride (halfs): 32 K + 8 pad keeps 16B alignment, spreads banks
#define LDK 40

__device__ __forceinline__ float sgnf(float t) {
    return (t > 0.f) ? 1.f : ((t < 0.f) ? -1.f : 0.f);
}

// A fragment: 16x32 f16.  lane L: row = base + (L&15); half hf = L>>4
//   regs 0..3  -> K = hf*8 .. hf*8+7        (contiguous 16B)
//   regs 4..7  -> K = 16 + hf*8 .. +7       (contiguous 16B)
__device__ __forceinline__ v16h frag_a(const _Float16* s, int row, int hf) {
    const h8 lo = *(const h8*)(s + row * LDK + hf * 8);
    const h8 hi = *(const h8*)(s + row * LDK + 16 + hf * 8);
    v16h r;
#pragma unroll
    for (int i = 0; i < 8; ++i) { r[i] = lo[i]; r[8 + i] = hi[i]; }
    return r;
}

// B fragment: 32x16 f16 (KxN).  lane L: col = base + (L&15); hf = L>>4
//   regs 0..3 -> K = hf*8 .. hf*8+7, regs 4..7 -> K = 16 + hf*8 .. +7 (same as A)
__device__ __forceinline__ v16h frag_b(const _Float16* s, int col, int hf) {
    const h8 lo = *(const h8*)(s + col * LDK + hf * 8);
    const h8 hi = *(const h8*)(s + col * LDK + 16 + hf * 8);
    v16h r;
#pragma unroll
    for (int i = 0; i < 8; ++i) { r[i] = lo[i]; r[8 + i] = hi[i]; }
    return r;
}

// ---------------------------------------------------------------------------
// Tile staging: A 64 rows x 32 K, B 128 rows(N) x 32 K.
// Each thread moves 1 A b128 + 2 B b128  (=> 3 outstanding async ops/stage).
// ---------------------------------------------------------------------------
#define ASYNC_OPS_PER_STAGE 3

__device__ __forceinline__ void stage_tiles(const _Float16* __restrict__ A,
                                            const _Float16* __restrict__ Wnk,
                                            _Float16* sA, _Float16* sB,
                                            int m0, int n0, int kk, int ldk,
                                            int t) {
    int ra = t >> 2, ka = (t & 3) << 3;
    const _Float16* ap = A + (size_t)(m0 + ra) * ldk + kk + ka;
    int rb = t >> 1, kb = (t & 1) << 4;
    const _Float16* wp = Wnk + (size_t)(n0 + rb) * ldk + kk + kb;
    _Float16* la = sA + ra * LDK + ka;
    _Float16* lb = sB + rb * LDK + kb;
#if USE_ASYNC_LDS
    __builtin_amdgcn_global_load_async_to_lds_b128(
        (v4i_vs*)ap, (v4i_vs*)la, 0, 0);
    __builtin_amdgcn_global_load_async_to_lds_b128(
        (v4i_vs*)wp, (v4i_vs*)lb, 0, 0);
    __builtin_amdgcn_global_load_async_to_lds_b128(
        (v4i_vs*)(wp + 8), (v4i_vs*)(lb + 8), 0, 0);
#else
    *(h8*)la       = *(const h8*)ap;
    *(h8*)lb       = *(const h8*)wp;
    *(h8*)(lb + 8) = *(const h8*)(wp + 8);
#endif
}

__device__ __forceinline__ void stage_wait(bool more_in_flight) {
#if USE_ASYNC_LDS
    // async loads complete in order per wave: waiting down to the count of
    // the next (in-flight) stage guarantees the current stage has landed.
    if (more_in_flight)
        __builtin_amdgcn_s_wait_asynccnt(ASYNC_OPS_PER_STAGE);
    else
        __builtin_amdgcn_s_wait_asynccnt(0);
#else
    (void)more_in_flight;
#endif
}

// ---------------------------------------------------------------------------
// K0: weight quantization.  One thread per output channel.
// ---------------------------------------------------------------------------
__global__ void k_quant_weights(const float* __restrict__ dw_w,
                                const float* __restrict__ pw1_w,
                                const float* __restrict__ pw2_w,
                                float* __restrict__ wq_dw,
                                _Float16* __restrict__ wq1,
                                _Float16* __restrict__ wq2) {
    int tid = blockIdx.x * blockDim.x + threadIdx.x;
    if (tid < DIM) {
        const float* w = dw_w + tid * 49;
        float s = 0.f;
        for (int i = 0; i < 49; ++i) s += w[i];
        float bw = s * (1.f / 49.f);
        float v = 0.f;
        for (int i = 0; i < 49; ++i) { float d = w[i] - bw; v += d * d; }
        float aw = sqrtf(v * (1.f / 49.f));
        for (int i = 0; i < 49; ++i)
            wq_dw[tid * 49 + i] = sgnf(w[i] - bw) * aw + bw;
    } else if (tid < DIM + EXP_) {
        int e = tid - DIM;
        const float* w = pw1_w + (size_t)e * DIM;
        float s = 0.f;
        for (int i = 0; i < DIM; ++i) s += w[i];
        float bw = s * (1.f / DIM);
        float v = 0.f;
        for (int i = 0; i < DIM; ++i) { float d = w[i] - bw; v += d * d; }
        float aw = sqrtf(v * (1.f / DIM));
        for (int i = 0; i < DIM; ++i)
            wq1[(size_t)e * DIM + i] = (_Float16)(sgnf(w[i] - bw) * aw + bw);
    } else if (tid < DIM + EXP_ + DIM) {
        int o = tid - DIM - EXP_;
        const float* w = pw2_w + (size_t)o * EXP_;
        float s = 0.f;
        for (int i = 0; i < EXP_; ++i) s += w[i];
        float bw = s * (1.f / EXP_);
        float v = 0.f;
        for (int i = 0; i < EXP_; ++i) { float d = w[i] - bw; v += d * d; }
        float aw = sqrtf(v * (1.f / EXP_));
        for (int i = 0; i < EXP_; ++i)
            wq2[(size_t)o * EXP_ + i] = (_Float16)(sgnf(w[i] - bw) * aw + bw);
    }
}

// ---------------------------------------------------------------------------
// K1: depthwise 7x7 (pad 3) on binarized input + bn1 + residual, then
// quantize with pw1 (alpha,beta) and store fp16 A-matrix h_q[p][c].
// One block per (n, c) plane; 256 thr x 4 pixels.
// ---------------------------------------------------------------------------
__global__ void k_dwconv(const float* __restrict__ x,
                         const float* __restrict__ wq_dw,
                         const float* __restrict__ dw_alpha,
                         const float* __restrict__ dw_beta,
                         const float* __restrict__ bn1_g,
                         const float* __restrict__ bn1_b,
                         const float* __restrict__ bn1_m,
                         const float* __restrict__ bn1_v,
                         const float* __restrict__ pw1_alpha,
                         const float* __restrict__ pw1_beta,
                         _Float16* __restrict__ h_q) {
    __shared__ float sw[49];
    int n = blockIdx.x / DIM;
    int c = blockIdx.x % DIM;
    int t = threadIdx.x;
    if (t < 49) sw[t] = wq_dw[c * 49 + t];
    __syncthreads();

    const float da  = dw_alpha[c], db = dw_beta[c];
    const float ira = 1.f / da;
    const float inv = bn1_g[c] * rsqrtf(bn1_v[c] + BN_EPS);
    const float add = bn1_b[c] - bn1_m[c] * inv;
    const float a1  = pw1_alpha[c], b1 = pw1_beta[c];
    const float ir1 = 1.f / a1;

    const float* xp = x + ((size_t)(n * DIM + c) << 10);

#pragma unroll
    for (int q = 0; q < 4; ++q) {
        int pix = t + (q << 8);
        int hh = pix >> 5, ww = pix & 31;
        float acc = 0.f;
#pragma unroll
        for (int kh = 0; kh < 7; ++kh) {
            int ih = hh - 3 + kh;
            if ((unsigned)ih >= 32u) continue;
#pragma unroll
            for (int kw = 0; kw < 7; ++kw) {
                int iw = ww - 3 + kw;
                if ((unsigned)iw >= 32u) continue;
                float xv = xp[(ih << 5) + iw];
                float xq = sgnf((xv - db) * ira) * da + db;  // zero-pad outside
                acc += xq * sw[kh * 7 + kw];
            }
        }
        float hv = xp[pix] + acc * inv + add;                // x + bn1(conv)
        float hq = sgnf((hv - b1) * ir1) * a1 + b1;          // pw1 act quant
        h_q[(size_t)((n << 10) + pix) * DIM + c] = (_Float16)hq;
    }
}

// ---------------------------------------------------------------------------
// K2: GEMM1  (32768 x 1536 x K=384) + bn2 + maxout + pw2-quant -> a2 fp16
// 256 thr = 8 waves; tile 64M x 128N; wave = 32x32 (2x2 wmma);
// double-buffered async LDS pipeline.
// ---------------------------------------------------------------------------
__global__ void __launch_bounds__(256)
k_gemm_expand(const _Float16* __restrict__ Aq,
              const _Float16* __restrict__ Wnk,
              _Float16* __restrict__ a2,
              const float* __restrict__ g2, const float* __restrict__ b2,
              const float* __restrict__ m2, const float* __restrict__ v2,
              const float* __restrict__ pos, const float* __restrict__ neg,
              const float* __restrict__ al2, const float* __restrict__ be2) {
    __shared__ _Float16 sA[2][64 * LDK];
    __shared__ _Float16 sB[2][128 * LDK];

    const int t = threadIdx.x;
    const int m0 = blockIdx.x * 64;
    const int n0 = blockIdx.y * 128;
    const int lane = t & 31, wid = t >> 5;
    const int mW = (wid >> 2) * 32, nW = (wid & 3) * 32;
    const int hf = lane >> 4, l16 = lane & 15;
    const int nk = DIM / 32;

    v8f acc[2][2] = {};

    stage_tiles(Aq, Wnk, sA[0], sB[0], m0, n0, 0, DIM, t);
    for (int i = 0; i < nk; ++i) {
        const int cur = i & 1;
        const bool more = (i + 1) < nk;
        if (more)
            stage_tiles(Aq, Wnk, sA[cur ^ 1], sB[cur ^ 1],
                        m0, n0, (i + 1) * 32, DIM, t);
        stage_wait(more);
        __syncthreads();

        v16h a0  = frag_a(sA[cur], mW + l16, hf);
        v16h a1  = frag_a(sA[cur], mW + 16 + l16, hf);
        v16h bb0 = frag_b(sB[cur], nW + l16, hf);
        v16h bb1 = frag_b(sB[cur], nW + 16 + l16, hf);

        acc[0][0] = __builtin_amdgcn_wmma_f32_16x16x32_f16(false, a0, false, bb0, (short)0, acc[0][0], false, false);
        acc[0][1] = __builtin_amdgcn_wmma_f32_16x16x32_f16(false, a0, false, bb1, (short)0, acc[0][1], false, false);
        acc[1][0] = __builtin_amdgcn_wmma_f32_16x16x32_f16(false, a1, false, bb0, (short)0, acc[1][0], false, false);
        acc[1][1] = __builtin_amdgcn_wmma_f32_16x16x32_f16(false, a1, false, bb1, (short)0, acc[1][1], false, false);
        __syncthreads();
    }

    // epilogue: bn2 + maxout + pw2 activation quantization, store fp16
#pragma unroll
    for (int j = 0; j < 2; ++j) {
        int col = n0 + nW + j * 16 + l16;
        float inv = g2[col] * rsqrtf(v2[col] + BN_EPS);
        float add = b2[col] - m2[col] * inv;
        float p = pos[col], q = neg[col];
        float al = al2[col], be = be2[col];
        float ial = 1.f / al;
#pragma unroll
        for (int i = 0; i < 2; ++i) {
            int mb = m0 + mW + i * 16 + hf * 8;
#pragma unroll
            for (int e = 0; e < 8; ++e) {
                float y = acc[i][j][e] * inv + add;
                y = p * fmaxf(y, 0.f) - q * fmaxf(-y, 0.f);
                float yq = sgnf((y - be) * ial) * al + be;
                a2[(size_t)(mb + e) * EXP_ + col] = (_Float16)yq;
            }
        }
    }
}

// ---------------------------------------------------------------------------
// K3: GEMM2  (32768 x 384 x K=1536) + bn3 + input residual -> fp32 NCHW
// ---------------------------------------------------------------------------
__global__ void __launch_bounds__(256)
k_gemm_project(const _Float16* __restrict__ Aq,
               const _Float16* __restrict__ Wnk,
               const float* __restrict__ x,
               float* __restrict__ out,
               const float* __restrict__ g3, const float* __restrict__ b3,
               const float* __restrict__ m3, const float* __restrict__ v3) {
    __shared__ _Float16 sA[2][64 * LDK];
    __shared__ _Float16 sB[2][128 * LDK];

    const int t = threadIdx.x;
    const int m0 = blockIdx.x * 64;
    const int n0 = blockIdx.y * 128;
    const int lane = t & 31, wid = t >> 5;
    const int mW = (wid >> 2) * 32, nW = (wid & 3) * 32;
    const int hf = lane >> 4, l16 = lane & 15;
    const int nk = EXP_ / 32;

    v8f acc[2][2] = {};

    stage_tiles(Aq, Wnk, sA[0], sB[0], m0, n0, 0, EXP_, t);
    for (int i = 0; i < nk; ++i) {
        const int cur = i & 1;
        const bool more = (i + 1) < nk;
        if (more)
            stage_tiles(Aq, Wnk, sA[cur ^ 1], sB[cur ^ 1],
                        m0, n0, (i + 1) * 32, EXP_, t);
        stage_wait(more);
        __syncthreads();

        v16h a0  = frag_a(sA[cur], mW + l16, hf);
        v16h a1  = frag_a(sA[cur], mW + 16 + l16, hf);
        v16h bb0 = frag_b(sB[cur], nW + l16, hf);
        v16h bb1 = frag_b(sB[cur], nW + 16 + l16, hf);

        acc[0][0] = __builtin_amdgcn_wmma_f32_16x16x32_f16(false, a0, false, bb0, (short)0, acc[0][0], false, false);
        acc[0][1] = __builtin_amdgcn_wmma_f32_16x16x32_f16(false, a0, false, bb1, (short)0, acc[0][1], false, false);
        acc[1][0] = __builtin_amdgcn_wmma_f32_16x16x32_f16(false, a1, false, bb0, (short)0, acc[1][0], false, false);
        acc[1][1] = __builtin_amdgcn_wmma_f32_16x16x32_f16(false, a1, false, bb1, (short)0, acc[1][1], false, false);
        __syncthreads();
    }

    // epilogue: bn3 + residual with original x, scatter to NCHW fp32
#pragma unroll
    for (int j = 0; j < 2; ++j) {
        int col = n0 + nW + j * 16 + l16;              // output channel
        float inv = g3[col] * rsqrtf(v3[col] + BN_EPS);
        float add = b3[col] - m3[col] * inv;
#pragma unroll
        for (int i = 0; i < 2; ++i) {
            int mb = m0 + mW + i * 16 + hf * 8;
#pragma unroll
            for (int e = 0; e < 8; ++e) {
                int mrow = mb + e;                     // spatial index p
                int nb = mrow >> 10;                   // batch
                int hw = mrow & 1023;
                size_t idx = ((size_t)(nb * DIM + col) << 10) + hw;
                out[idx] = x[idx] + acc[i][j][e] * inv + add;
            }
        }
    }
}

// ---------------------------------------------------------------------------
// Host launcher
// ---------------------------------------------------------------------------
extern "C" void kernel_launch(void* const* d_in, const int* in_sizes, int n_in,
                              void* d_out, int out_size, void* d_ws, size_t ws_size,
                              hipStream_t stream) {
    (void)in_sizes; (void)n_in; (void)out_size; (void)ws_size;

    const float* x         = (const float*)d_in[0];
    const float* dw_w      = (const float*)d_in[1];
    const float* dw_alpha  = (const float*)d_in[2];
    const float* dw_beta   = (const float*)d_in[3];
    const float* bn1_g     = (const float*)d_in[4];
    const float* bn1_b     = (const float*)d_in[5];
    const float* bn1_m     = (const float*)d_in[6];
    const float* bn1_v     = (const float*)d_in[7];
    const float* pw1_w     = (const float*)d_in[8];
    const float* pw1_alpha = (const float*)d_in[9];
    const float* pw1_beta  = (const float*)d_in[10];
    const float* bn2_g     = (const float*)d_in[11];
    const float* bn2_b     = (const float*)d_in[12];
    const float* bn2_m     = (const float*)d_in[13];
    const float* bn2_v     = (const float*)d_in[14];
    const float* mx_pos    = (const float*)d_in[15];
    const float* mx_neg    = (const float*)d_in[16];
    const float* pw2_w     = (const float*)d_in[17];
    const float* pw2_alpha = (const float*)d_in[18];
    const float* pw2_beta  = (const float*)d_in[19];
    const float* bn3_g     = (const float*)d_in[20];
    const float* bn3_b     = (const float*)d_in[21];
    const float* bn3_m     = (const float*)d_in[22];
    const float* bn3_v     = (const float*)d_in[23];
    float* out = (float*)d_out;

    // workspace layout (16B-aligned offsets)
    char* ws = (char*)d_ws;
    constexpr size_t OFF_WQDW = 0;                               //  75,264 B fp32
    constexpr size_t OFF_WQ1  = 75264;                           // fp16 [1536][384]
    constexpr size_t OFF_WQ2  = OFF_WQ1 + 1179648;               // fp16 [384][1536]
    constexpr size_t OFF_HQ   = OFF_WQ2 + 1179648;               // fp16 [32768][384]
    constexpr size_t OFF_A2   = OFF_HQ + (size_t)M_TOT * DIM * 2;// fp16 [32768][1536]

    float*    wq_dw = (float*)(ws + OFF_WQDW);
    _Float16* wq1   = (_Float16*)(ws + OFF_WQ1);
    _Float16* wq2   = (_Float16*)(ws + OFF_WQ2);
    _Float16* h_q   = (_Float16*)(ws + OFF_HQ);
    _Float16* a2    = (_Float16*)(ws + OFF_A2);

    // K0: quantize weights (2304 channel-threads)
    k_quant_weights<<<9, 256, 0, stream>>>(dw_w, pw1_w, pw2_w, wq_dw, wq1, wq2);

    // K1: depthwise + bn1 + residual + pw1 activation quant
    k_dwconv<<<BATCH * DIM, 256, 0, stream>>>(x, wq_dw, dw_alpha, dw_beta,
                                              bn1_g, bn1_b, bn1_m, bn1_v,
                                              pw1_alpha, pw1_beta, h_q);

    // K2: expand GEMM + bn2 + maxout + pw2 activation quant
    k_gemm_expand<<<dim3(M_TOT / 64, EXP_ / 128), 256, 0, stream>>>(
        h_q, wq1, a2, bn2_g, bn2_b, bn2_m, bn2_v,
        mx_pos, mx_neg, pw2_alpha, pw2_beta);

    // K3: project GEMM + bn3 + input residual
    k_gemm_project<<<dim3(M_TOT / 64, DIM / 128), 256, 0, stream>>>(
        a2, wq2, x, out, bn3_g, bn3_b, bn3_m, bn3_v);
}
